// GATv2_4707284156950
// MI455X (gfx1250) — matrix operand order repaired and checked by reading, not compiled, not run on hardware
//
#include <hip/hip_runtime.h>

// ---------------------------------------------------------------------------
// Types for CDNA5 WMMA
// ---------------------------------------------------------------------------
typedef __attribute__((ext_vector_type(16))) __bf16 v16bf;
typedef __attribute__((ext_vector_type(8)))  float  v8f;

union ABFrag {
    uint4 u[2];   // 32 bytes = 16 bf16
    v16bf v;
};

__device__ __forceinline__ unsigned short f32_to_bf16(float f) {
    unsigned u = __float_as_uint(f);
    unsigned r = u + 0x7FFFu + ((u >> 16) & 1u);   // round-to-nearest-even
    return (unsigned short)(r >> 16);
}

// order-preserving float->uint key for atomicMax-based segment max
__device__ __forceinline__ unsigned f32_to_key(float f) {
    unsigned u = __float_as_uint(f);
    return (u & 0x80000000u) ? ~u : (u | 0x80000000u);
}
__device__ __forceinline__ float key_to_f32(unsigned k) {
    unsigned u = (k & 0x80000000u) ? (k ^ 0x80000000u) : ~k;
    return __uint_as_float(u);
}

// ---------------------------------------------------------------------------
// fp32 -> bf16 conversion of activations
// ---------------------------------------------------------------------------
__global__ void cvt_x_kernel(const float* __restrict__ x,
                             unsigned short* __restrict__ xb, int n) {
    int i = blockIdx.x * blockDim.x + threadIdx.x;
    if (i < n) xb[i] = f32_to_bf16(x[i]);
}

// transpose 256x256 weight to [n][k] bf16 so B-fragment K runs are contiguous
__global__ void cvt_w_kernel(const float* __restrict__ W,
                             unsigned short* __restrict__ WT) {
    int k = threadIdx.x;   // 0..255
    int n = blockIdx.x;    // 0..255
    WT[n * 256 + k] = f32_to_bf16(W[k * 256 + n]);
}

// ---------------------------------------------------------------------------
// WMMA GEMM: Y[M x 256] = bf16(X[M x 256]) @ bf16(W[256 x 256]) + bias
// 512 threads = 16 waves; wave w computes the 16x16 tile at (blockIdx.x*16, w*16)
// K-loop: 8 x v_wmma_f32_16x16x32_bf16, fp32 accumulation.
// Fragment layouts per CDNA5 ISA 7.12.2 (16-bit A 16x32, B 32x16, f32 C/D).
// ---------------------------------------------------------------------------
__global__ __launch_bounds__(512)
void gemm_bf16_256(const unsigned short* __restrict__ X,
                   const unsigned short* __restrict__ WT,  // [n][k]
                   const float* __restrict__ bias,
                   float* __restrict__ Y) {
    const int lane  = threadIdx.x & 31;
    const int wave  = threadIdx.x >> 5;          // 0..15
    const int m0    = blockIdx.x * 16;
    const int n0    = wave * 16;
    const int row   = m0 + (lane & 15);          // A: lanes 0-15 / 16-31 both carry M=0..15
    const int koffA = (lane < 16) ? 0 : 8;       // A: lo lanes K 0-7 & 16-23, hi lanes 8-15 & 24-31
    const int ncol  = n0 + (lane & 15);          // B/C/D: N striped over lanes
    const int koffB = (lane < 16) ? 0 : 8;       // B: lo lanes K 0-7 & 16-23, hi lanes 8-15 & 24-31

    const unsigned short* xrow = X  + (size_t)row  * 256;
    const unsigned short* wrow = WT + (size_t)ncol * 256;

    v8f acc = {};
#pragma unroll
    for (int k = 0; k < 256; k += 32) {
        ABFrag a, b;
        a.u[0] = *(const uint4*)(xrow + k + koffA);        // K = k+koff   .. +7
        a.u[1] = *(const uint4*)(xrow + k + 16 + koffA);   // K = k+16+koff.. +7
        b.u[0] = *(const uint4*)(wrow + k + koffB);        // K = k+koffB  .. +7
        b.u[1] = *(const uint4*)(wrow + k + 16 + koffB);   // K = k+16+koffB.. +7
        acc = __builtin_amdgcn_wmma_f32_16x16x32_bf16(
            /*neg_a=*/false, a.v, /*neg_b=*/false, b.v,
            /*c_mod=*/(short)0, acc, /*reuse_a=*/false, /*reuse_b=*/false);
    }

    const float bv = bias[ncol];
    const int r0 = m0 + ((lane < 16) ? 0 : 8);   // C/D: VGPR j -> M = j (+8 for hi lanes)
    float* yp = Y + (size_t)r0 * 256 + ncol;
#pragma unroll
    for (int j = 0; j < 8; ++j) yp[(size_t)j * 256] = acc[j] + bv;
}

// ---------------------------------------------------------------------------
// Per-call re-init of softmax stats + output accumulator (graph-replay safe)
// ---------------------------------------------------------------------------
__global__ void init_kernel(unsigned* __restrict__ mkey, float* __restrict__ den,
                            float* __restrict__ Hacc, int n4, int nH) {
    int i = blockIdx.x * blockDim.x + threadIdx.x;
    if (i < n4) { mkey[i] = 0u; den[i] = 0.0f; }   // key 0 < key(any finite float)
    if (i < nH) Hacc[i] = 0.0f;
}

// ---------------------------------------------------------------------------
// Edge phase 1: logits[e][h] = att_h . leaky_relu(xl[src] + xr[dst]); seg-max
// One wave per edge: lane l covers channels [8l, 8l+8), head = l>>3.
// ---------------------------------------------------------------------------
__global__ __launch_bounds__(256)
void edge_logits_kernel(const float* __restrict__ XL, const float* __restrict__ XR,
                        const int* __restrict__ esrc, const int* __restrict__ edst,
                        const float* __restrict__ att,
                        float* __restrict__ logits, unsigned* __restrict__ mkey,
                        int E, int E2) {
    int e = blockIdx.x * 8 + (threadIdx.x >> 5);
    if (e >= E2) return;
    int lane = threadIdx.x & 31;
    int s = (e < E) ? esrc[e] : (e - E);
    int d = (e < E) ? edst[e] : (e - E);
    int c0 = lane * 8;
    const float4* pl = (const float4*)(XL + (size_t)s * 256 + c0);
    const float4* pr = (const float4*)(XR + (size_t)d * 256 + c0);
    const float4* pa = (const float4*)(att + c0);
    float partial = 0.0f;
#pragma unroll
    for (int q = 0; q < 2; ++q) {
        float4 a = pl[q], b = pr[q], w = pa[q];
        float v;
        v = a.x + b.x; v = v > 0.0f ? v : 0.2f * v; partial += w.x * v;
        v = a.y + b.y; v = v > 0.0f ? v : 0.2f * v; partial += w.y * v;
        v = a.z + b.z; v = v > 0.0f ? v : 0.2f * v; partial += w.z * v;
        v = a.w + b.w; v = v > 0.0f ? v : 0.2f * v; partial += w.w * v;
    }
    // reduce across the 8 lanes of each head
    partial += __shfl_xor(partial, 1, 32);
    partial += __shfl_xor(partial, 2, 32);
    partial += __shfl_xor(partial, 4, 32);
    if ((lane & 7) == 0) {
        int h = lane >> 3;
        logits[(size_t)e * 4 + h] = partial;
        atomicMax(&mkey[d * 4 + h], f32_to_key(partial));
    }
}

// ---------------------------------------------------------------------------
// Edge phase 2: ex = exp(logit - max); segment-sum denominators
// ---------------------------------------------------------------------------
__global__ void edge_exp_kernel(const int* __restrict__ edst,
                                const unsigned* __restrict__ mkey,
                                float* __restrict__ logits, float* __restrict__ den,
                                int E, int E2) {
    int i = blockIdx.x * blockDim.x + threadIdx.x;
    if (i >= E2 * 4) return;
    int e = i >> 2, h = i & 3;
    int d = (e < E) ? edst[e] : (e - E);
    float m = key_to_f32(mkey[d * 4 + h]);
    float ex = __expf(logits[i] - m);
    logits[i] = ex;
    atomicAdd(&den[d * 4 + h], ex);
}

// ---------------------------------------------------------------------------
// Edge phase 3: H[dst] += alpha * xl[src]  (one wave per edge, float atomics
// into the 20 MB accumulator — L2 resident on a 192 MB L2)
// ---------------------------------------------------------------------------
__global__ __launch_bounds__(256)
void edge_msg_kernel(const float* __restrict__ XL,
                     const int* __restrict__ esrc, const int* __restrict__ edst,
                     const float* __restrict__ exv, const float* __restrict__ den,
                     float* __restrict__ Hacc, int E, int E2) {
    int e = blockIdx.x * 8 + (threadIdx.x >> 5);
    if (e >= E2) return;
    int lane = threadIdx.x & 31;
    int s = (e < E) ? esrc[e] : (e - E);
    int d = (e < E) ? edst[e] : (e - E);
    int h = lane >> 3;
    float alpha = exv[(size_t)e * 4 + h] / den[d * 4 + h];
    int c0 = lane * 8;
    const float4* px = (const float4*)(XL + (size_t)s * 256 + c0);
    float4 x0 = px[0], x1 = px[1];
    float* o = Hacc + (size_t)d * 256 + c0;
    atomicAdd(o + 0, alpha * x0.x);
    atomicAdd(o + 1, alpha * x0.y);
    atomicAdd(o + 2, alpha * x0.z);
    atomicAdd(o + 3, alpha * x0.w);
    atomicAdd(o + 4, alpha * x1.x);
    atomicAdd(o + 5, alpha * x1.y);
    atomicAdd(o + 6, alpha * x1.z);
    atomicAdd(o + 7, alpha * x1.w);
}

// ---------------------------------------------------------------------------
// Bias + ReLU. mode 0: write bf16 back as next layer's GEMM input.
//             mode 1: write fp32 final output.
// ---------------------------------------------------------------------------
__global__ void finalize_kernel(const float* __restrict__ Hacc,
                                const float* __restrict__ bias,
                                unsigned short* __restrict__ xb_out,
                                float* __restrict__ f_out, int n, int mode) {
    int i = blockIdx.x * blockDim.x + threadIdx.x;
    if (i >= n) return;
    float v = Hacc[i] + bias[i & 255];
    v = v > 0.0f ? v : 0.0f;
    if (mode == 0) xb_out[i] = f32_to_bf16(v);
    else           f_out[i] = v;
}

// ---------------------------------------------------------------------------
// Host orchestration
// ---------------------------------------------------------------------------
extern "C" void kernel_launch(void* const* d_in, const int* in_sizes, int n_in,
                              void* d_out, int out_size, void* d_ws, size_t ws_size,
                              hipStream_t stream) {
    const int N = 20000, E = 320000, E2 = E + N, HID = 256;

    const float* x    = (const float*)d_in[0];
    const int*   eidx = (const int*)d_in[1];
    const int*   esrc = eidx;        // edge_index row 0
    const int*   edst = eidx + E;    // edge_index row 1

    // workspace layout (256B aligned slabs)
    char* ws = (char*)d_ws;
    size_t off = 0;
    auto slab = [&](size_t bytes) -> char* {
        char* p = ws + off;
        off += (bytes + 255) & ~(size_t)255;
        return p;
    };
    unsigned short* XBF  = (unsigned short*)slab((size_t)N * HID * 2);  // bf16 layer input
    unsigned short* WTL  = (unsigned short*)slab(256 * 256 * 2);        // bf16 W_l^T
    unsigned short* WTR  = (unsigned short*)slab(256 * 256 * 2);        // bf16 W_r^T
    float*    XL   = (float*)slab((size_t)N * HID * 4);                 // x @ Wl + bl
    float*    XR   = (float*)slab((size_t)N * HID * 4);                 // x @ Wr + br
    float*    LOG  = (float*)slab((size_t)E2 * 4 * 4);                  // logits -> exp
    unsigned* MKEY = (unsigned*)slab((size_t)N * 4 * 4);                // seg-max keys
    float*    DEN  = (float*)slab((size_t)N * 4 * 4);                   // seg-sum
    float*    HACC = (float*)slab((size_t)N * HID * 4);                 // output accumulator
    (void)ws_size;

    const int nElem = N * HID;
    cvt_x_kernel<<<(nElem + 255) / 256, 256, 0, stream>>>(x, XBF, nElem);

    for (int li = 0; li < 2; ++li) {
        const float* Wl   = (const float*)d_in[2 + li * 6 + 0];
        const float* bl   = (const float*)d_in[2 + li * 6 + 1];
        const float* Wr   = (const float*)d_in[2 + li * 6 + 2];
        const float* br   = (const float*)d_in[2 + li * 6 + 3];
        const float* att  = (const float*)d_in[2 + li * 6 + 4];
        const float* bias = (const float*)d_in[2 + li * 6 + 5];

        cvt_w_kernel<<<256, 256, 0, stream>>>(Wl, WTL);
        cvt_w_kernel<<<256, 256, 0, stream>>>(Wr, WTR);

        gemm_bf16_256<<<N / 16, 512, 0, stream>>>(XBF, WTL, bl, XL);
        gemm_bf16_256<<<N / 16, 512, 0, stream>>>(XBF, WTR, br, XR);

        init_kernel<<<(nElem + 255) / 256, 256, 0, stream>>>(MKEY, DEN, HACC, N * 4, nElem);

        edge_logits_kernel<<<(E2 + 7) / 8, 256, 0, stream>>>(XL, XR, esrc, edst, att,
                                                             LOG, MKEY, E, E2);
        edge_exp_kernel<<<(E2 * 4 + 255) / 256, 256, 0, stream>>>(edst, MKEY, LOG, DEN, E, E2);
        edge_msg_kernel<<<(E2 + 7) / 8, 256, 0, stream>>>(XL, esrc, edst, LOG, DEN, HACC, E, E2);

        finalize_kernel<<<(nElem + 255) / 256, 256, 0, stream>>>(HACC, bias, XBF,
                                                                 (float*)d_out, nElem, li);
    }
}
